// SetConv_45380624449643
// MI455X (gfx1250) — matrix operand order repaired and checked by reading, not compiled, not run on hardware
//
#include <hip/hip_runtime.h>
#include <hip/hip_bf16.h>

// ---------------------------------------------------------------------------
// PointNet++ SetConv for MI455X (gfx1250, wave32, WMMA).
//
// Pipeline (all on `stream`, graph-capture safe):
//   1) copy q -> d_out[0:49152]
//   2) ball_query -> idx (B*M*K int32)            [VALU scan, early exit]
//   3) W_i f32 -> f16 padded (Cout x Kpad)        [tiny]
//   4) feat0 build: gather + relcoords -> X f16 (P x 96, zero padded)
//   5) per layer: WMMA GEMM (f16 in, f32 acc) -> Y (P x Cout)
//                 stats (sum/sumsq per channel) -> finalize (scale/shift)
//                 BN+ReLU+cast f16 -> X (next layer input)  [layers 0,1]
//   6) fused BN+ReLU+maxpool over K -> d_out[49152:]
//
// Train-mode BN: bias b cancels ((Wx+b) - mean(Wx+b) == Wx - mean(Wx)),
// so only (gamma, beta, mean, var) are folded into scale/shift.
// ---------------------------------------------------------------------------

typedef _Float16 v16h __attribute__((ext_vector_type(16)));
typedef _Float16 v8h  __attribute__((ext_vector_type(8)));
typedef float    v8f  __attribute__((ext_vector_type(8)));

#define Bc   16
#define Nc   4096
#define Mc   1024
#define Cc   64
#define Kc   32
#define Pc   (Bc * Mc * Kc)      // 524288 gathered points
#define R2c  (0.2f * 0.2f)
#define QOUT (Bc * Mc * 3)       // 49152 floats of q in d_out

// ---------------------------------------------------------------------------
// 1) copy q into the first output slot
__global__ void k_copy(const float* __restrict__ src, float* __restrict__ dst, int n) {
    int t = blockIdx.x * blockDim.x + threadIdx.x;
    if (t < n) dst[t] = src[t];
}

// ---------------------------------------------------------------------------
// 2) ball query: first K indices (in index order) with d2 < R^2, padded with
//    the first valid index (0 if none). One thread per (b,m); lanes in a wave
//    share b so p[] reads broadcast out of L0/L2.
__global__ void k_ball_query(const float* __restrict__ p, const float* __restrict__ q,
                             int* __restrict__ idx) {
    int t = blockIdx.x * blockDim.x + threadIdx.x;
    if (t >= Bc * Mc) return;
    int b = t / Mc, m = t - b * Mc;
    const float* qp = q + (size_t)t * 3;
    float qx = qp[0], qy = qp[1], qz = qp[2];
    const float* pb = p + (size_t)b * Nc * 3;
    int* out = idx + (size_t)t * Kc;
    int cnt = 0, first = -1;
    for (int n = 0; n < Nc; ++n) {
        float dx = pb[n * 3 + 0] - qx;
        float dy = pb[n * 3 + 1] - qy;
        float dz = pb[n * 3 + 2] - qz;
        float d2 = dx * dx + dy * dy + dz * dz;
        if (d2 < R2c) {
            if (first < 0) first = n;
            out[cnt++] = n;
            if (cnt == Kc) break;
        }
    }
    if (first < 0) first = 0;
    for (; cnt < Kc; ++cnt) out[cnt] = first;
}

// ---------------------------------------------------------------------------
// 3) weights f32 (Cout x Cin) -> f16 (Cout x Kpad) zero padded
__global__ void k_convert_w(const float* __restrict__ W, _Float16* __restrict__ Wh,
                            int Cout, int Cin, int Kpad) {
    int t = blockIdx.x * blockDim.x + threadIdx.x;
    if (t >= Cout * Kpad) return;
    int o = t / Kpad, c = t - o * Kpad;
    Wh[t] = (_Float16)(c < Cin ? W[o * Cin + c] : 0.0f);
}

// ---------------------------------------------------------------------------
// 4) feature build: X0[pt*96 + c] f16, pt = (b*M+m)*K + kk
//    c in [0,3):   p[b,n,c] - q[b,m,c]
//    c in [3,67):  x[b, c-3, n]
//    c in [67,96): 0 (K-dim padding so WMMA K-loop is exact)
__global__ void k_build_feat(const float* __restrict__ p, const float* __restrict__ q,
                             const float* __restrict__ x, const int* __restrict__ idx,
                             _Float16* __restrict__ X0) {
    int pt = blockIdx.x;            // 0 .. P-1
    int c  = threadIdx.x;           // 0 .. 95
    int bm = pt / Kc;
    int b  = bm / Mc, m = bm - b * Mc;
    int n  = idx[pt];
    float v = 0.0f;
    if (c < 3) {
        v = p[((size_t)b * Nc + n) * 3 + c] - q[((size_t)b * Mc + m) * 3 + c];
    } else if (c < 3 + Cc) {
        v = x[((size_t)b * Cc + (c - 3)) * (size_t)Nc + n];
    }
    X0[(size_t)pt * 96 + c] = (_Float16)v;
}

// ---------------------------------------------------------------------------
// 5a) WMMA GEMM: Y(P x Cout) = Wh(Cout x Kpad) * X(P x Kpad)^T, f32 accumulate.
//     One wave per 16(Cout) x 16(points) tile; block = Cout/16 waves sharing a
//     point tile. Operand layouts per CDNA5 ISA 7.12.2 (wave32):
//       A f16 16x32: lane l -> row l&15; halves 0..7 = K (l>=16 ? 8 : 0)..+7,
//                    halves 8..15 = +16  -> two contiguous 16B loads
//       B f16 32x16: lane l -> col l&15; same K order as A
//                    -> two contiguous 16B loads (point-major features)
//       C/D f32:     lane l -> col l&15, rows 8*(l>=16)..+7 -> 8 contiguous
//                    floats of the (P x Cout) output row
__global__ void k_gemm_wmma(const _Float16* __restrict__ X,
                            const _Float16* __restrict__ Wh,
                            float* __restrict__ Y,
                            int Kpad, int Cout) {
    const int lane  = threadIdx.x & 31;
    const int tileM = threadIdx.x >> 5;        // which 16-row band of Cout
    const int r     = lane & 15;
    const int hi    = lane >> 4;               // 0 or 1
    const _Float16* aRow = Wh + (size_t)(tileM * 16 + r) * Kpad + hi * 8;
    const _Float16* bCol = X  + ((size_t)blockIdx.x * 16 + r) * Kpad + hi * 8;

    v8f acc = {};
    for (int k = 0; k < Kpad; k += 32) {
        v16h a, bfrag;
        *(v8h*)&a         = *(const v8h*)(aRow + k);        // K halves 0..7
        *((v8h*)&a + 1)   = *(const v8h*)(aRow + k + 16);   // K halves 8..15
        *(v8h*)&bfrag       = *(const v8h*)(bCol + k);      // K halves 0..7
        *((v8h*)&bfrag + 1) = *(const v8h*)(bCol + k + 16); // K halves 8..15
        acc = __builtin_amdgcn_wmma_f32_16x16x32_f16(
            /*neg_a=*/false, a, /*neg_b=*/false, bfrag,
            /*c_mod=*/(short)0, acc, /*reuse_a=*/false, /*reuse_b=*/false);
    }

    float* yp = Y + ((size_t)blockIdx.x * 16 + r) * Cout + tileM * 16 + hi * 8;
#pragma unroll
    for (int j = 0; j < 8; ++j) yp[j] = acc[j];
}

// ---------------------------------------------------------------------------
// 5b) per-channel sum / sumsq over Y (P x Cout), Cout is a power of two and
//     divides blockDim, so element channel == threadIdx.x % Cout throughout
//     the grid stride. LDS reduce, then few atomics per block.
__global__ void k_stats(const float* __restrict__ Y, size_t total, int Cout,
                        float* __restrict__ sums /* [0..127]=sum, [128..255]=sumsq */) {
    __shared__ float ls[512];
    int c = threadIdx.x & (Cout - 1);
    float s = 0.0f, s2 = 0.0f;
    size_t stride = (size_t)gridDim.x * blockDim.x;
    for (size_t i = (size_t)blockIdx.x * blockDim.x + threadIdx.x; i < total; i += stride) {
        float v = Y[i];
        s += v; s2 += v * v;
    }
    ls[threadIdx.x] = s;
    ls[256 + threadIdx.x] = s2;
    __syncthreads();
    if ((int)threadIdx.x < Cout) {
        for (int j = threadIdx.x + Cout; j < 256; j += Cout) {
            s += ls[j]; s2 += ls[256 + j];
        }
        atomicAdd(&sums[c], s);
        atomicAdd(&sums[128 + c], s2);
    }
}

__global__ void k_zero(float* __restrict__ ptr, int n) {
    int t = blockIdx.x * blockDim.x + threadIdx.x;
    if (t < n) ptr[t] = 0.0f;
}

// 5c) fold (mean, var, gamma, beta) -> (scale, shift)
__global__ void k_finalize(const float* __restrict__ sums, const float* __restrict__ g,
                           const float* __restrict__ be, float* __restrict__ scale,
                           float* __restrict__ shift, int Cout, float invCount) {
    int c = threadIdx.x;
    if (c >= Cout) return;
    float mean = sums[c] * invCount;
    float var  = sums[128 + c] * invCount - mean * mean;
    float s    = g[c] * rsqrtf(var + 1e-5f);
    scale[c] = s;
    shift[c] = be[c] - mean * s;
}

// 5d) BN + ReLU + cast to f16 next-layer input (point-major, Kpad == Cout)
__global__ void k_bn_relu_cast(const float* __restrict__ Y,
                               const float* __restrict__ scale,
                               const float* __restrict__ shift,
                               _Float16* __restrict__ Xo, size_t total, int Cout) {
    size_t t = (size_t)blockIdx.x * blockDim.x + threadIdx.x;
    if (t >= total) return;
    int c = (int)(t & (size_t)(Cout - 1));
    float v = Y[t] * scale[c] + shift[c];
    Xo[t] = (_Float16)(v > 0.0f ? v : 0.0f);
}

// ---------------------------------------------------------------------------
// 6) fused BN + ReLU + maxpool over K for the last layer.
//    out layout (B, 128, M) at d_out + QOUT. max(relu(.)) >= 0 so init 0.
__global__ void k_pool(const float* __restrict__ Y,
                       const float* __restrict__ scale,
                       const float* __restrict__ shift,
                       float* __restrict__ out) {
    int t = blockIdx.x * blockDim.x + threadIdx.x;
    if (t >= Bc * 128 * Mc) return;
    int m = t % Mc;
    int o = (t / Mc) & 127;
    int b = t / (Mc * 128);
    const float* y = Y + ((size_t)(b * Mc + m) * Kc) * 128 + o;
    float s = scale[o], sh = shift[o];
    float mx = 0.0f;
    for (int k = 0; k < Kc; ++k) {
        float v = y[(size_t)k * 128] * s + sh;
        mx = fmaxf(mx, v > 0.0f ? v : 0.0f);
    }
    out[t] = mx;
}

// ---------------------------------------------------------------------------
static inline size_t align256(size_t v) { return (v + 255) & ~(size_t)255; }

extern "C" void kernel_launch(void* const* d_in, const int* in_sizes, int n_in,
                              void* d_out, int out_size, void* d_ws, size_t ws_size,
                              hipStream_t stream) {
    (void)in_sizes; (void)n_in; (void)out_size; (void)ws_size;

    const float* p  = (const float*)d_in[0];
    const float* q  = (const float*)d_in[1];
    const float* x  = (const float*)d_in[2];
    const float* W0 = (const float*)d_in[3];
    const float* g0 = (const float*)d_in[5];
    const float* be0= (const float*)d_in[6];
    const float* W1 = (const float*)d_in[7];
    const float* g1 = (const float*)d_in[9];
    const float* be1= (const float*)d_in[10];
    const float* W2 = (const float*)d_in[11];
    const float* g2 = (const float*)d_in[13];
    const float* be2= (const float*)d_in[14];
    float* out = (float*)d_out;

    // ---- workspace carve-up --------------------------------------------
    char* ws = (char*)d_ws;
    size_t off = 0;
    int* idx      = (int*)(ws + off);          off = align256(off + (size_t)Bc * Mc * Kc * sizeof(int));
    _Float16* Wh0 = (_Float16*)(ws + off);     off = align256(off + (size_t)64  * 96 * sizeof(_Float16));
    _Float16* Wh1 = (_Float16*)(ws + off);     off = align256(off + (size_t)64  * 64 * sizeof(_Float16));
    _Float16* Wh2 = (_Float16*)(ws + off);     off = align256(off + (size_t)128 * 64 * sizeof(_Float16));
    float* sums   = (float*)(ws + off);        off = align256(off + 256 * sizeof(float));
    float* scale  = (float*)(ws + off);        off = align256(off + 128 * sizeof(float));
    float* shift  = (float*)(ws + off);        off = align256(off + 128 * sizeof(float));
    _Float16* X   = (_Float16*)(ws + off);     off = align256(off + (size_t)Pc * 96 * sizeof(_Float16));
    float* Y      = (float*)(ws + off);        off = align256(off + (size_t)Pc * 128 * sizeof(float));

    const float invP = 1.0f / (float)Pc;       // BN count per channel = B*M*K

    // 1) q passthrough
    k_copy<<<(QOUT + 255) / 256, 256, 0, stream>>>(q, out, QOUT);

    // 2) ball query
    k_ball_query<<<(Bc * Mc + 255) / 256, 256, 0, stream>>>(p, q, idx);

    // 3) weight conversion
    k_convert_w<<<(64  * 96 + 255) / 256, 256, 0, stream>>>(W0, Wh0, 64, 67, 96);
    k_convert_w<<<(64  * 64 + 255) / 256, 256, 0, stream>>>(W1, Wh1, 64, 64, 64);
    k_convert_w<<<(128 * 64 + 255) / 256, 256, 0, stream>>>(W2, Wh2, 128, 64, 64);

    // 4) gather + feature build (P x 96 f16)
    k_build_feat<<<Pc, 96, 0, stream>>>(p, q, x, idx, X);

    const int tilesN = Pc / 16;                // 32768 point tiles

    // ---- layer 0: 96 -> 64 ---------------------------------------------
    k_gemm_wmma<<<tilesN, 4 * 32, 0, stream>>>(X, Wh0, Y, 96, 64);
    k_zero<<<1, 256, 0, stream>>>(sums, 256);
    k_stats<<<1024, 256, 0, stream>>>(Y, (size_t)Pc * 64, 64, sums);
    k_finalize<<<1, 128, 0, stream>>>(sums, g0, be0, scale, shift, 64, invP);
    k_bn_relu_cast<<<(unsigned)(((size_t)Pc * 64 + 255) / 256), 256, 0, stream>>>(
        Y, scale, shift, X, (size_t)Pc * 64, 64);

    // ---- layer 1: 64 -> 64 ---------------------------------------------
    k_gemm_wmma<<<tilesN, 4 * 32, 0, stream>>>(X, Wh1, Y, 64, 64);
    k_zero<<<1, 256, 0, stream>>>(sums, 256);
    k_stats<<<1024, 256, 0, stream>>>(Y, (size_t)Pc * 64, 64, sums);
    k_finalize<<<1, 128, 0, stream>>>(sums, g1, be1, scale, shift, 64, invP);
    k_bn_relu_cast<<<(unsigned)(((size_t)Pc * 64 + 255) / 256), 256, 0, stream>>>(
        Y, scale, shift, X, (size_t)Pc * 64, 64);

    // ---- layer 2: 64 -> 128 --------------------------------------------
    k_gemm_wmma<<<tilesN, 8 * 32, 0, stream>>>(X, Wh2, Y, 64, 128);
    k_zero<<<1, 256, 0, stream>>>(sums, 256);
    k_stats<<<1024, 256, 0, stream>>>(Y, (size_t)Pc * 128, 128, sums);
    k_finalize<<<1, 128, 0, stream>>>(sums, g2, be2, scale, shift, 128, invP);

    // 6) BN + ReLU + maxpool over K -> (B,128,M)
    k_pool<<<(Bc * 128 * Mc + 255) / 256, 256, 0, stream>>>(Y, scale, shift, out + QOUT);
}
